// QuantPillarVFE_15899968930293
// MI455X (gfx1250) — matrix operand order repaired and checked by reading, not compiled, not run on hardware
//
#include <hip/hip_runtime.h>

typedef __attribute__((ext_vector_type(16))) _Float16 v16h;
typedef __attribute__((ext_vector_type(8)))  _Float16 h8;
typedef __attribute__((ext_vector_type(8)))  float    v8f;

union V16 { v16h v; h8 h[2]; };

__device__ __forceinline__ v8f wmma16(v16h a, v16h b, v8f c) {
    // D = A(16x32 f16) * B(32x16 f16) + C(16x16 f32)
    return __builtin_amdgcn_wmma_f32_16x16x32_f16(
        false, a, false, b, (short)0, c, false, false);
}

// quant level of relu(x) given PRE-SCALED s = x/0.05:
// clamp(round(s), 0, 255). Mul-first keeps inputs canonical -> no v_max(x,x).
__device__ __forceinline__ float fq_level_scaled(float s) {
    float q = rintf(fmaxf(s, 0.0f));     // v_rndne_f32, half-even == jnp.round
    return fminf(q, 255.0f);
}

__device__ __forceinline__ v8f bcast8(float s) {
    v8f r;
    #pragma unroll
    for (int i = 0; i < 8; ++i) r[i] = s;
    return r;
}

__global__ __launch_bounds__(256) void pfn_wmma_kernel(
    const float* __restrict__ vf,      // [M,32,4]
    const float* __restrict__ W1,      // [32,10]  (out, in)
    const float* __restrict__ B1,      // [32]
    const float* __restrict__ W2,      // [128,64] (out, in)
    const float* __restrict__ B2,      // [128]
    const int*   __restrict__ npts,    // [M]
    const int*   __restrict__ coords,  // [M,4]
    float* __restrict__ out,           // [M,128]
    int M)
{
    // ---- LDS: shared weights + per-wave staging (8 waves/block) ----
    __shared__ __align__(16) _Float16 w1s[32 * 32];     //  2 KB, K padded 10->32
    __shared__ __align__(16) _Float16 w2s[128 * 64];    // 16 KB, pre-scaled by 0.05
    __shared__ float b1s[32];
    __shared__ float b2s[128];
    __shared__ __align__(16) _Float16 feat[8][32 * 32]; // per-point 10-dim (pad 32) f16
    __shared__ __align__(16) _Float16 hbuf[8][32 * 32]; // layer-1 quant levels (f16 ints)
    __shared__ __align__(16) _Float16 hmaxs[8][32];     // per-voxel channel max

    const int tid  = threadIdx.x;
    const int lane = tid & 31;
    const int wid  = tid >> 5;
    const int l15  = lane & 15;
    const int hs   = lane >> 4;          // which 16-lane half of the wave

    // ---- stage weights/biases to LDS once per block ----
    for (int i = tid; i < 32 * 32; i += 256) {
        int o = i >> 5, k = i & 31;
        w1s[i] = (_Float16)((k < 10) ? W1[o * 10 + k] : 0.0f);
    }
    // fold the layer-1 dequant scale (0.05) into W2 so hbuf holds raw quant levels
    for (int i = tid; i < 128 * 64; i += 256)
        w2s[i] = (_Float16)(W2[i] * 0.05f);
    if (tid < 32)  b1s[tid] = B1[tid];
    if (tid < 128) b2s[tid] = B2[tid];
    __syncthreads();

    const int stride = gridDim.x * 8;    // voxels per grid pass (wave-uniform)

    for (int v = blockIdx.x * 8 + wid; v < M; v += stride) {
        // prefetch next voxel tile into caches (global_prefetch_b8)
        if (v + stride < M)
            __builtin_prefetch(vf + (size_t)(v + stride) * 128 + lane * 4, 0, 1);

        // ---------- build 10-dim features; lane == point ----------
        const float4 p4 = ((const float4*)vf)[(size_t)v * 32 + lane];
        float x = p4.x, y = p4.y, z = p4.z, w = p4.w;

        // mean over ALL 32 stored points / n (matches reference: sum before mask)
        float sx = x, sy = y, sz = z;
        #pragma unroll
        for (int off = 16; off > 0; off >>= 1) {
            sx += __shfl_xor(sx, off, 32);
            sy += __shfl_xor(sy, off, 32);
            sz += __shfl_xor(sz, off, 32);
        }
        int n = npts[v]; n = (n < 1) ? 1 : n;
        const float rinv = 1.0f / (float)n;
        const float mx = sx * rinv, my = sy * rinv, mz = sz * rinv;

        const float cx = (float)coords[v * 4 + 3];
        const float cy = (float)coords[v * 4 + 2];
        const float cz = (float)coords[v * 4 + 1];
        const float fcx = x - (cx * 0.16f + 0.08f);
        const float fcy = y - (cy * 0.16f + 0.08f);
        const float fcz = z - (cz * 4.0f  - 1.0f);

        const float msk = (lane < n) ? 1.0f : 0.0f;   // value select: EXEC stays full
        {
            h8 r0 = { (_Float16)(x * msk),  (_Float16)(y * msk),
                      (_Float16)(z * msk),  (_Float16)(w * msk),
                      (_Float16)((x - mx) * msk), (_Float16)((y - my) * msk),
                      (_Float16)((z - mz) * msk), (_Float16)(fcx * msk) };
            h8 r1 = { (_Float16)(fcy * msk), (_Float16)(fcz * msk),
                      (_Float16)0, (_Float16)0, (_Float16)0,
                      (_Float16)0, (_Float16)0, (_Float16)0 };
            h8 zz = {};
            h8* rowp = (h8*)&feat[wid][lane * 32];
            rowp[0] = r0; rowp[1] = r1; rowp[2] = zz; rowp[3] = zz;
        }

        // A-fragment: row-major [32-half] rows; ISA 16-bit A layout
        // (lanes 0-15: K 0-7 & 16-23, lanes 16-31: K 8-15 & 24-31)
        auto loadA = [&](const _Float16* base, int row) -> v16h {
            V16 u;
            const _Float16* p = base + row * 32 + hs * 8;
            u.h[0] = *(const h8*)p;
            u.h[1] = *(const h8*)(p + 16);
            return u.v;
        };
        // B-fragment from [N][Kpad] weights; ISA 16-bit B layout
        // (lanes 0-15: K 0-7 & 16-23, lanes 16-31: K 8-15 & 24-31)
        auto loadB = [&](const _Float16* wsb, int Kpad, int t, int ks) -> v16h {
            V16 u;
            const _Float16* p = wsb + (t * 16 + l15) * Kpad + ks * 32 + hs * 8;
            u.h[0] = *(const h8*)p;
            u.h[1] = *(const h8*)(p + 16);
            return u.v;
        };

        // ---------- layer 1: (32x10) @ W1^T + b1 -> quant levels (32x32) ----------
        const _Float16* fb = &feat[wid][0];
        const v8f cb0 = bcast8(b1s[l15]);        // bias folded into C operand
        const v8f cb1 = bcast8(b1s[16 + l15]);
        {
            v16h a0 = loadA(fb, l15);            // points 0-15
            v16h a1 = loadA(fb, 16 + l15);       // points 16-31
            v16h bW0 = loadB(w1s, 32, 0, 0);     // out channels 0-15
            v16h bW1 = loadB(w1s, 32, 1, 0);     // out channels 16-31
            // pre-scale by 1/0.05: v_pk_mul, outputs canonical for max chains
            v8f s00 = wmma16(a0, bW0, cb0) * 20.0f;
            v8f s01 = wmma16(a0, bW1, cb1) * 20.0f;
            v8f s10 = wmma16(a1, bW0, cb0) * 20.0f;
            v8f s11 = wmma16(a1, bW1, cb1) * 20.0f;

            _Float16* hb = &hbuf[wid][0];
            float hm0 = 0.0f, hm1 = 0.0f;
            #pragma unroll
            for (int r = 0; r < 8; ++r) {
                const int p0 = r + 8 * hs;       // C layout: VGPR r = rows r / r+8
                const int p1 = 16 + r + 8 * hs;
                float h00 = fq_level_scaled(s00[r]);
                float h01 = fq_level_scaled(s01[r]);
                float h10 = fq_level_scaled(s10[r]);
                float h11 = fq_level_scaled(s11[r]);
                hb[p0 * 32 + l15]      = (_Float16)h00;   // exact: integer <= 255
                hb[p0 * 32 + 16 + l15] = (_Float16)h01;
                hb[p1 * 32 + l15]      = (_Float16)h10;
                hb[p1 * 32 + 16 + l15] = (_Float16)h11;
                hm0 = fmaxf(fmaxf(hm0, h00), h10);
                hm1 = fmaxf(fmaxf(hm1, h01), h11);
            }
            hm0 = fmaxf(hm0, __shfl_xor(hm0, 16, 32));
            hm1 = fmaxf(hm1, __shfl_xor(hm1, 16, 32));
            if (lane < 16) {
                hmaxs[wid][lane]      = (_Float16)hm0;
                hmaxs[wid][16 + lane] = (_Float16)hm1;
            }
        }

        // ---------- layer 2: concat([h, hmax]) (32x64) @ (0.05*W2)^T -> max ----------
        const _Float16* hb = &hbuf[wid][0];
        v16h A0 = loadA(hb, l15);                // K-step 0, points 0-15
        v16h A1 = loadA(hb, 16 + l15);           // K-step 0, points 16-31
        V16 uh;                                  // K-step 1: broadcast hmax rows
        uh.h[0] = *(const h8*)(&hmaxs[wid][0] + hs * 8);
        uh.h[1] = *(const h8*)(&hmaxs[wid][0] + 16 + hs * 8);
        const v16h AH = uh.v;

        const size_t ob = (size_t)v * 128;
        #pragma unroll 2                         // keep VGPRs < 256
        for (int t = 0; t < 8; ++t) {
            const v8f cb = bcast8(b2s[t * 16 + l15]);   // bias via C operand
            v16h B0  = loadB(w2s, 64, t, 0);
            v16h B1t = loadB(w2s, 64, t, 1);
            // AH rows are identical for both M-tiles: compute its product ONCE,
            // then reuse as the C operand of both row-tiles (3 WMMAs, not 4).
            v8f e  = wmma16(AH, B1t, cb);
            v8f d0 = wmma16(A0, B0, e);
            v8f d1 = wmma16(A1, B0, e);

            // fq is monotone: reduce first, quantize once. Pre-scale via v_pk_mul
            // so the 15-deep max chain runs on canonical values (no v_max(x,x)).
            v8f t0 = d0 * 20.0f;
            v8f t1 = d1 * 20.0f;
            float m = t0[0];
            #pragma unroll
            for (int r = 1; r < 8; ++r) m = fmaxf(m, t0[r]);
            #pragma unroll
            for (int r = 0; r < 8; ++r) m = fmaxf(m, t1[r]);
            m = fmaxf(m, __shfl_xor(m, 16, 32)); // rows 0-7,16-23 vs 8-15,24-31
            if (lane < 16)
                out[ob + t * 16 + lane] = fq_level_scaled(m) * 0.05f;
        }
    }
}

extern "C" void kernel_launch(void* const* d_in, const int* in_sizes, int n_in,
                              void* d_out, int out_size, void* d_ws, size_t ws_size,
                              hipStream_t stream) {
    const float* vf     = (const float*)d_in[0];
    const float* W1     = (const float*)d_in[1];
    const float* B1     = (const float*)d_in[2];
    const float* W2     = (const float*)d_in[3];
    const float* B2     = (const float*)d_in[4];
    const int*   npts   = (const int*)d_in[5];
    const int*   coords = (const int*)d_in[6];
    float*       outp   = (float*)d_out;

    const int M = in_sizes[5];                 // 160000 voxels
    int blocks = (M + 7) / 8;
    if (blocks > 1024) blocks = 1024;          // grid-stride: amortize LDS weight staging
    pfn_wmma_kernel<<<blocks, 256, 0, stream>>>(vf, W1, B1, W2, B2, npts, coords, outp, M);
}
